// NodeFormer_29781303230688
// MI455X (gfx1250) — matrix operand order repaired and checked by reading, not compiled, not run on hardware
//
#include <hip/hip_runtime.h>
#include <math.h>

typedef float  v8f     __attribute__((ext_vector_type(8)));
typedef __bf16 bf16x16 __attribute__((ext_vector_type(16)));

#define H     4
#define D     64
#define MREAL 30
#define MP    32
#define KK    10
#define CIN   128
#define HD    256
#define KM    320   // KK*MP

__device__ __constant__ float SQ    = 0.7071067811865476f;   // (1/sqrt(tau)) * 64^{-1/4}
__device__ __constant__ float RATIO = 0.18257418583505536f;  // 30^{-1/2}
__device__ __constant__ float EPSF  = 1e-6f;
__device__ __constant__ float INVK  = 0.1f;

__device__ __forceinline__ int fragKA(int lane, int j){ return j + 16*(lane>>4); }
__device__ __forceinline__ int fragKB(int lane, int j){ return j + 16*(lane>>4); }

__device__ __forceinline__ v8f vzero(){ v8f r = {0.f,0.f,0.f,0.f,0.f,0.f,0.f,0.f}; return r; }

__device__ __forceinline__ v8f wmma_bf16(bf16x16 a, bf16x16 b, v8f c){
  return __builtin_amdgcn_wmma_f32_16x16x32_bf16(false, a, false, b, (short)0, c, false, false);
}

__device__ __forceinline__ bf16x16 load_bfrag(const __bf16* frags, int ct, int ks, int KS, int lane){
  const __bf16* p = frags + ((size_t)((ct*KS + ks)*32 + lane))*16;
  bf16x16 b;
  #pragma unroll
  for (int j = 0; j < 16; ++j) b[j] = p[j];
  return b;
}

// monotone float<->uint key for atomic max
__device__ __forceinline__ unsigned fenc(float f){
  unsigned u = __float_as_uint(f);
  return (u & 0x80000000u) ? ~u : (u | 0x80000000u);
}
__device__ __forceinline__ float fdec(unsigned k){
  return (k & 0x80000000u) ? __uint_as_float(k & 0x7fffffffu) : __uint_as_float(~k);
}

// ---------------------------------------------------------------- kernel 0
// zero accumulators; repack Wq/Wk/Wv/projT/Wo into bf16 B-fragment order
__global__ void k0_init_repack(const float* wq, const float* wk, const float* wv,
                               const float* proj, const float* wo,
                               float* kvs, float* ksum, float* kpsum, unsigned* gmax,
                               __bf16* wqf, __bf16* wkf, __bf16* wvf,
                               __bf16* projf, __bf16* wof)
{
  int t = blockIdx.x*blockDim.x + threadIdx.x;
  if (t < H*KM*D)  { kvs[t]   = 0.f; return; }  t -= H*KM*D;
  if (t < H*KK*MP) { ksum[t]  = 0.f; return; }  t -= H*KK*MP;
  if (t < H*MP)    { kpsum[t] = 0.f; return; }  t -= H*MP;
  if (t < H)       { gmax[t]  = 0u;  return; }  t -= H;
  if (t < 3*32768){                                   // Wq/Wk/Wv: 16 ct x 4 ks
    const float* W = (t < 32768) ? wq : (t < 65536 ? wk : wv);
    __bf16* dst    = (t < 32768) ? wqf : (t < 65536 ? wkf : wvf);
    int i = t & 32767;
    int j = i & 15, L = (i>>4)&31, ks = (i>>9)&3, ct = i>>11;
    dst[i] = (__bf16)W[(ks*32 + fragKB(L,j))*HD + ct*16 + (L&15)];
    return;
  } t -= 3*32768;
  if (t < 2048){                                      // projT: 2 ct x 2 ks (B[d][m]=proj[m][d])
    int j = t & 15, L = (t>>4)&31, ks = (t>>9)&1, ct = t>>10;
    int d = ks*32 + fragKB(L,j), m = ct*16 + (L&15);
    projf[t] = (__bf16)((m < MREAL) ? proj[m*D + d] : 0.f);
    return;
  } t -= 2048;
  if (t < 16384){                                     // Wo [256][64]: 4 ct x 8 ks
    int j = t & 15, L = (t>>4)&31, ks = (t>>9)&7, ct = t>>12;
    wof[t] = (__bf16)wo[(ks*32 + fragKB(L,j))*D + ct*16 + (L&15)];
  }
}

// ---------------------------------------------------------------- kernel 1 helper
__device__ __forceinline__ void qkv_tiles(const __bf16 (*zs)[CIN], const __bf16* wf,
                                          const float* bias, float scale, int mode,
                                          __bf16 (*xb)[HD], __bf16* vbf,
                                          int nbase, int N, int lane, int w)
{
  for (int t = w; t < 64; t += 8){
    int rt = t >> 4, ct = t & 15;
    v8f acc = vzero();
    int r = rt*16 + (lane&15);
    #pragma unroll
    for (int ks = 0; ks < 4; ++ks){
      bf16x16 a;
      #pragma unroll
      for (int j = 0; j < 16; ++j) a[j] = zs[r][ks*32 + fragKA(lane,j)];
      acc = wmma_bf16(a, load_bfrag(wf, ct, ks, 4, lane), acc);
    }
    int col = ct*16 + (lane&15);
    float bv = bias[col];
    #pragma unroll
    for (int r8 = 0; r8 < 8; ++r8){
      int nl = rt*16 + r8 + 8*(lane>>4);
      float val = acc[r8] + bv;
      if (mode == 2){
        int gn = nbase + nl;
        if (gn < N) vbf[(size_t)gn*HD + col] = (__bf16)val;
      } else {
        xb[nl][col] = (__bf16)(val * scale);
      }
    }
  }
}

// ---------------------------------------------------------------- kernel 1
// fused QKV projection + Performer feature maps; 64 nodes per block, 8 waves
__global__ void __launch_bounds__(256)
k1_qkv_feat(const float* z, const float* wqb, const float* wkb, const float* wvb,
            const __bf16* wqf, const __bf16* wkf, const __bf16* wvf,
            const __bf16* projf,
            __bf16* vbf, float* qpg, float* skg, unsigned* gmax, int N)
{
  __shared__ __bf16 zs[64][CIN];     // 16 KB
  __shared__ __bf16 xb[64][HD];      // 32 KB (reused: q then k)
  __shared__ float  diagl[64][H];    // 1 KB
  __shared__ unsigned lmax[H];
  const int tid = threadIdx.x, lane = tid & 31, w = tid >> 5;
  const int nbase = blockIdx.x * 64;

  if (tid < H) lmax[tid] = 0u;
  for (int i = tid; i < 64*CIN; i += 256){
    int n = i >> 7, c = i & 127, gn = nbase + n;
    zs[n][c] = (gn < N) ? (__bf16)z[(size_t)gn*CIN + c] : (__bf16)0.f;
  }
  __syncthreads();

  // ---- phase Q ----
  qkv_tiles(zs, wqf, wqb, SQ, 0, xb, nullptr, nbase, N, lane, w);
  __syncthreads();
  { int n = tid >> 2, h = tid & 3; float s = 0.f;
    #pragma unroll
    for (int d = 0; d < D; ++d){ float xv = (float)xb[n][h*D + d]; s += xv*xv; }
    diagl[n][h] = 0.5f * s; }
  __syncthreads();
  for (int pp = w; pp < 16; pp += 8){
    int h = pp >> 2, rt = pp & 3;
    v8f a0 = vzero(), a1 = vzero();
    int r = rt*16 + (lane&15);
    #pragma unroll
    for (int ks = 0; ks < 2; ++ks){
      bf16x16 a;
      #pragma unroll
      for (int j = 0; j < 16; ++j) a[j] = xb[r][h*D + ks*32 + fragKA(lane,j)];
      a0 = wmma_bf16(a, load_bfrag(projf, 0, ks, 2, lane), a0);
      a1 = wmma_bf16(a, load_bfrag(projf, 1, ks, 2, lane), a1);
    }
    bool c1ok = (lane & 15) < (MREAL - 16);
    #pragma unroll
    for (int r8 = 0; r8 < 8; ++r8){
      float mx = fmaxf(a0[r8], c1ok ? a1[r8] : -__builtin_inff());
      #pragma unroll
      for (int off = 1; off < 16; off <<= 1) mx = fmaxf(mx, __shfl_xor(mx, off, 16));
      int nl = rt*16 + r8 + 8*(lane>>4), gn = nbase + nl;
      float dv = diagl[nl][h];
      float q0 = RATIO * (expf(a0[r8] - dv - mx) + EPSF);
      float q1 = c1ok ? RATIO * (expf(a1[r8] - dv - mx) + EPSF) : 0.f;
      if (gn < N){
        float* qrow = qpg + ((size_t)gn*H + h)*MP;
        qrow[lane & 15] = q0;
        qrow[16 + (lane & 15)] = q1;
      }
    }
  }
  __syncthreads();

  // ---- phase K ----
  qkv_tiles(zs, wkf, wkb, SQ, 1, xb, nullptr, nbase, N, lane, w);
  __syncthreads();
  { int n = tid >> 2, h = tid & 3; float s = 0.f;
    #pragma unroll
    for (int d = 0; d < D; ++d){ float xv = (float)xb[n][h*D + d]; s += xv*xv; }
    diagl[n][h] = 0.5f * s; }
  __syncthreads();
  for (int pp = w; pp < 16; pp += 8){
    int h = pp >> 2, rt = pp & 3;
    v8f a0 = vzero(), a1 = vzero();
    int r = rt*16 + (lane&15);
    #pragma unroll
    for (int ks = 0; ks < 2; ++ks){
      bf16x16 a;
      #pragma unroll
      for (int j = 0; j < 16; ++j) a[j] = xb[r][h*D + ks*32 + fragKA(lane,j)];
      a0 = wmma_bf16(a, load_bfrag(projf, 0, ks, 2, lane), a0);
      a1 = wmma_bf16(a, load_bfrag(projf, 1, ks, 2, lane), a1);
    }
    bool c1ok = (lane & 15) < (MREAL - 16);
    float lm = -__builtin_inff();
    #pragma unroll
    for (int r8 = 0; r8 < 8; ++r8){
      int nl = rt*16 + r8 + 8*(lane>>4), gn = nbase + nl;
      if (gn < N){
        float dv = diagl[nl][h];
        float* srow = skg + ((size_t)gn*H + h)*MP;
        srow[lane & 15] = a0[r8] - dv;
        srow[16 + (lane & 15)] = a1[r8] - dv;
        lm = fmaxf(lm, fmaxf(a0[r8], c1ok ? a1[r8] : -__builtin_inff()));
      }
    }
    #pragma unroll
    for (int off = 1; off < 32; off <<= 1) lm = fmaxf(lm, __shfl_xor(lm, off, 32));
    if (lane == 0) atomicMax(&lmax[h], fenc(lm));
  }
  __syncthreads();
  if (tid < H) atomicMax(&gmax[tid], lmax[tid]);

  // ---- phase V ----
  qkv_tiles(zs, wvf, wvb, 1.f, 2, xb, vbf, nbase, N, lane, w);
}

// ---------------------------------------------------------------- kernel 2
// kp = ratio*(exp(sk - gmax)+eps); kvs[h] += Wᵀ·V (reduction over N), one head/block
__global__ void __launch_bounds__(256)
k2_kvs(const float* skg, const float* gum, const __bf16* vbf,
       const unsigned* gmax, float* kpg, float* kvsg,
       float* ksumg, float* kpsumg, int N, int nsplit)
{
  __shared__ __bf16 wl[64][KM];     // 40 KB  wl[n][k*32+m] = kp*exp(g)
  __shared__ __bf16 vl[64][D];      // 8 KB
  __shared__ float  egl[64][KK];    // 2.5 KB
  __shared__ float  kss[KK*MP];
  __shared__ float  kps[MP];
  const int tid = threadIdx.x, lane = tid & 31, w = tid >> 5;
  const int h = blockIdx.y;
  const float mx = fdec(gmax[h]);
  if (tid < KK*MP) kss[tid] = 0.f;
  if (tid < MP)    kps[tid] = 0.f;
  const int mcol = tid & 31;
  float ksr[KK];
  #pragma unroll
  for (int k = 0; k < KK; ++k) ksr[k] = 0.f;
  float kpr = 0.f;
  v8f acc[10];
  #pragma unroll
  for (int ti = 0; ti < 10; ++ti) acc[ti] = vzero();

  const int nch = (N + 63) >> 6;
  for (int c = blockIdx.x; c < nch; c += nsplit){
    const int base = c * 64;
    __syncthreads();
    // prefetch next chunk's streams into cache while this chunk computes
    { int nc = c + nsplit;
      if (nc < nch && tid < 64){
        int gn2 = nc*64 + tid;
        if (gn2 < N){
          __builtin_prefetch(skg + ((size_t)gn2*H + h)*MP, 0, 1);
          __builtin_prefetch(vbf + (size_t)gn2*HD + h*D, 0, 1);
          __builtin_prefetch(gum + (size_t)gn2*(H*KK) + h*KK, 0, 1);
        }
      }
    }
    for (int i = tid; i < 64*KK; i += 256){
      int n = i / KK, k = i - n*KK, gn = base + n;
      egl[n][k] = (gn < N) ? expf(gum[(size_t)gn*(H*KK) + h*KK + k]) : 0.f;
    }
    for (int i = tid; i < 64*D; i += 256){
      int n = i >> 6, d = i & 63, gn = base + n;
      vl[n][d] = (gn < N) ? vbf[(size_t)gn*HD + h*D + d] : (__bf16)0.f;
    }
    __syncthreads();
    #pragma unroll
    for (int p = 0; p < 8; ++p){
      int n = (tid + p*256) >> 5, gn = base + n;
      float kp = 0.f;
      if (gn < N && mcol < MREAL)
        kp = RATIO * (expf(skg[((size_t)gn*H + h)*MP + mcol] - mx) + EPSF);
      if (gn < N) kpg[((size_t)gn*H + h)*MP + mcol] = kp;
      kpr += kp;
      #pragma unroll
      for (int k = 0; k < KK; ++k){
        float val = kp * egl[n][k];
        ksr[k] += val;
        wl[n][k*MP + mcol] = (__bf16)val;
      }
    }
    __syncthreads();
    #pragma unroll
    for (int ti = 0; ti < 10; ++ti){
      int t = w + ti*8;
      int kmt = t >> 2, dt = t & 3;
      #pragma unroll
      for (int ks = 0; ks < 2; ++ks){
        bf16x16 a, b;
        #pragma unroll
        for (int j = 0; j < 16; ++j) a[j] = wl[ks*32 + fragKA(lane,j)][kmt*16 + (lane&15)];
        #pragma unroll
        for (int j = 0; j < 16; ++j) b[j] = vl[ks*32 + fragKB(lane,j)][dt*16 + (lane&15)];
        acc[ti] = wmma_bf16(a, b, acc[ti]);
      }
    }
  }
  #pragma unroll
  for (int ti = 0; ti < 10; ++ti){
    int t = w + ti*8;
    int kmt = t >> 2, dt = t & 3;
    #pragma unroll
    for (int r8 = 0; r8 < 8; ++r8){
      int km = kmt*16 + r8 + 8*(lane>>4), d = dt*16 + (lane&15);
      atomicAdd(&kvsg[((size_t)h*KM + km)*D + d], acc[ti][r8]);
    }
  }
  __syncthreads();
  #pragma unroll
  for (int k = 0; k < KK; ++k) atomicAdd(&kss[k*MP + mcol], ksr[k]);
  atomicAdd(&kps[mcol], kpr);
  __syncthreads();
  if (tid < KK*MP) atomicAdd(&ksumg[h*KK*MP + tid], kss[tid]);
  if (tid < MP)    atomicAdd(&kpsumg[h*MP + tid],  kps[tid]);
}

// ---------------------------------------------------------------- kernel 2b
// repack kvs f32 -> bf16 B-fragments: ((h*KK+k)*4 + dt) frags of 512
__global__ void k2b_pack(const float* kvsg, __bf16* kvsf){
  int t = blockIdx.x*blockDim.x + threadIdx.x;
  if (t >= H*KK*4*512) return;
  int j = t & 15, L = (t>>4)&31, dt = (t>>9)&3;
  int r = t >> 11, k = r % KK, h = r / KK;
  kvsf[t] = (__bf16)kvsg[((size_t)h*KM + k*MP + fragKB(L,j))*D + dt*16 + (L&15)];
}

// ---------------------------------------------------------------- kernel 3
// z_num WMMA + per-k 1/z_den + mean over K, fused output projection Wo
__global__ void __launch_bounds__(256)
k3_attn_out(const float* qpg, const float* ksumg, const __bf16* kvsf,
            const __bf16* wof, const float* wob, float* outp, int N)
{
  __shared__ __bf16 qpl[64][H*MP];   // 16 KB
  __shared__ float  zden[64][H*KK];  // 10 KB
  __shared__ __bf16 zol[64][HD];     // 32 KB
  const int tid = threadIdx.x, lane = tid & 31, w = tid >> 5;
  const int nbase = blockIdx.x * 64;

  for (int i = tid; i < 64*128; i += 256){
    int n = i >> 7, hm = i & 127, gn = nbase + n;
    qpl[n][hm] = (gn < N) ? (__bf16)qpg[(size_t)gn*128 + hm] : (__bf16)0.f;
  }
  __syncthreads();
  for (int i = tid; i < 64*H*KK; i += 256){
    int n = i / (H*KK), hk = i - n*(H*KK);
    int h = hk / KK, k = hk - h*KK;
    float s = 0.f;
    #pragma unroll
    for (int m = 0; m < MP; ++m)
      s += (float)qpl[n][h*MP + m] * ksumg[(h*KK + k)*MP + m];
    zden[n][hk] = s;
  }
  __syncthreads();

  const int rt = w & 3, dtg = w >> 2;
  for (int h = 0; h < H; ++h){
    bf16x16 a;
    #pragma unroll
    for (int j = 0; j < 16; ++j) a[j] = qpl[rt*16 + (lane&15)][h*MP + fragKA(lane,j)];
    v8f zo0 = vzero(), zo1 = vzero();
    for (int k = 0; k < KK; ++k){
      float rd[8];
      #pragma unroll
      for (int r8 = 0; r8 < 8; ++r8)
        rd[r8] = 1.f / zden[rt*16 + r8 + 8*(lane>>4)][h*KK + k];
      const __bf16* pb = kvsf + ((size_t)(h*KK + k)*4)*512;
      bf16x16 b0, b1;
      const __bf16* p0 = pb + ((size_t)((dtg*2 + 0)*32 + lane))*16;
      const __bf16* p1 = pb + ((size_t)((dtg*2 + 1)*32 + lane))*16;
      #pragma unroll
      for (int j = 0; j < 16; ++j){ b0[j] = p0[j]; b1[j] = p1[j]; }
      v8f t0 = wmma_bf16(a, b0, vzero());
      v8f t1 = wmma_bf16(a, b1, vzero());
      #pragma unroll
      for (int r8 = 0; r8 < 8; ++r8){ zo0[r8] += t0[r8]*rd[r8]; zo1[r8] += t1[r8]*rd[r8]; }
    }
    #pragma unroll
    for (int r8 = 0; r8 < 8; ++r8){
      int nl = rt*16 + r8 + 8*(lane>>4);
      zol[nl][h*D + (dtg*2 + 0)*16 + (lane&15)] = (__bf16)(zo0[r8]*INVK);
      zol[nl][h*D + (dtg*2 + 1)*16 + (lane&15)] = (__bf16)(zo1[r8]*INVK);
    }
  }
  __syncthreads();

  for (int cc = 0; cc < 2; ++cc){
    int ct = dtg*2 + cc;
    v8f acc = vzero();
    #pragma unroll
    for (int ks = 0; ks < 8; ++ks){
      bf16x16 a;
      #pragma unroll
      for (int j = 0; j < 16; ++j) a[j] = zol[rt*16 + (lane&15)][ks*32 + fragKA(lane,j)];
      acc = wmma_bf16(a, load_bfrag(wof, ct, ks, 8, lane), acc);
    }
    int col = ct*16 + (lane&15);
    float bv = wob[col];
    #pragma unroll
    for (int r8 = 0; r8 < 8; ++r8){
      int gn = nbase + rt*16 + r8 + 8*(lane>>4);
      if (gn < N) outp[(size_t)gn*D + col] = acc[r8] + bv;
    }
  }
}

// ---------------------------------------------------------------- kernel 4
// edge attention weights (qp/kp are L2-resident gathers)
__global__ void __launch_bounds__(256)
k4_edges(const int* ei, const float* qpg, const float* kpg,
         const float* kpsumg, float* aout, int E)
{
  int e = blockIdx.x*blockDim.x + threadIdx.x;
  if (e >= E) return;
  int s = ei[e], t = ei[E + e];
  #pragma unroll
  for (int h = 0; h < H; ++h){
    const float* qrow = qpg + ((size_t)t*H + h)*MP;
    const float* krow = kpg + ((size_t)s*H + h)*MP;
    const float* prow = kpsumg + h*MP;
    float num = 0.f, den = 0.f;
    #pragma unroll
    for (int m = 0; m < MP; ++m){ float qv = qrow[m]; num += qv*krow[m]; den += qv*prow[m]; }
    aout[(size_t)e*H + h] = num/den;
  }
}

// ---------------------------------------------------------------- launch
extern "C" void kernel_launch(void* const* d_in, const int* in_sizes, int n_in,
                              void* d_out, int out_size, void* d_ws, size_t ws_size,
                              hipStream_t stream)
{
  const float* z    = (const float*)d_in[0];
  const int*   ei   = (const int*)  d_in[1];
  const float* gum  = (const float*)d_in[2];
  const float* proj = (const float*)d_in[3];
  const float* wqw  = (const float*)d_in[4];
  const float* wqb  = (const float*)d_in[5];
  const float* wkw  = (const float*)d_in[6];
  const float* wkb  = (const float*)d_in[7];
  const float* wvw  = (const float*)d_in[8];
  const float* wvb  = (const float*)d_in[9];
  const float* wow  = (const float*)d_in[10];
  const float* wob  = (const float*)d_in[11];
  const int N = in_sizes[0] / CIN;
  const int E = in_sizes[1] / 2;
  float* outp = (float*)d_out;
  float* aout = outp + (size_t)N*D;

  char* p = (char*)d_ws;
  auto take = [&](size_t bytes)->char*{
    char* r = p; p += (bytes + 255) & ~(size_t)255; return r;
  };
  __bf16*   vbf   = (__bf16*)  take((size_t)N*HD*2);
  float*    qpg   = (float*)   take((size_t)N*H*MP*4);
  float*    skg   = (float*)   take((size_t)N*H*MP*4);
  float*    kpg   = (float*)   take((size_t)N*H*MP*4);
  float*    kvsg  = (float*)   take((size_t)H*KM*D*4);
  float*    ksumg = (float*)   take((size_t)H*KK*MP*4);
  float*    kpsg  = (float*)   take((size_t)H*MP*4);
  unsigned* gmax  = (unsigned*)take(H*4);
  __bf16*   wqf   = (__bf16*)  take(32768*2);
  __bf16*   wkf   = (__bf16*)  take(32768*2);
  __bf16*   wvf   = (__bf16*)  take(32768*2);
  __bf16*   projf = (__bf16*)  take(2048*2);
  __bf16*   wof   = (__bf16*)  take(16384*2);
  __bf16*   kvsf  = (__bf16*)  take((size_t)H*KK*4*512*2);

  const int t0 = H*KM*D + H*KK*MP + H*MP + H + 3*32768 + 2048 + 16384;
  k0_init_repack<<<(t0 + 255)/256, 256, 0, stream>>>(
      wqw, wkw, wvw, proj, wow, kvsg, ksumg, kpsg, gmax, wqf, wkf, wvf, projf, wof);

  const int nblk = (N + 63) / 64;
  k1_qkv_feat<<<nblk, 256, 0, stream>>>(
      z, wqb, wkb, wvb, wqf, wkf, wvf, projf, vbf, qpg, skg, gmax, N);

  const int NSPLIT = 96;
  k2_kvs<<<dim3(NSPLIT, H), 256, 0, stream>>>(
      skg, gum, vbf, gmax, kpg, kvsg, ksumg, kpsg, N, NSPLIT);

  k2b_pack<<<(H*KK*4*512 + 255)/256, 256, 0, stream>>>(kvsg, kvsf);

  k3_attn_out<<<nblk, 256, 0, stream>>>(qpg, ksumg, kvsf, wof, wob, outp, N);

  k4_edges<<<(E + 255)/256, 256, 0, stream>>>(ei, qpg, kpg, kpsg, aout, E);
}
